// FourierWeightsBlock_74045236183682
// MI455X (gfx1250) — compile-verified, matrix operand order checked by reading, not run on hardware
//
#include <hip/hip_runtime.h>
#include <hip/hip_bf16.h>

// MI455X / gfx1250, wave32. Fully fused two-branch MLP:
//   out = (cos(x@W1r * s_in + b) @ W2r * s_h) * (sigmoid(x@W1s * s_in) @ W2s * s_h)
// bf16 WMMA (v_wmma_f32_16x16x32_bf16), f32 accumulation, weights pre-packed
// fragment-major in d_ws, hidden activations staged in LDS (never hit HBM).

typedef __bf16 bf16_t;
typedef __attribute__((ext_vector_type(16))) __bf16 v16bf;
typedef __attribute__((ext_vector_type(8)))  float  v8f;

#define DIN    512
#define HDIM   2048
#define DOUT   512
#define MTILE  32          // rows per workgroup
#define CHUNK  64          // hidden columns processed per iteration
#define NCHUNK (HDIM / CHUNK)   // 32

#define S_IN  0.04419417382415922f    // 512^-0.5
#define S_H2  4.8828125e-4f           // (2048^-0.5)^2 folded into one scale

// ---------------------------------------------------------------------------
// One-time repack: fp32 [K][N] row-major  ->  bf16 WMMA-B fragments.
// Fragment = one 32x16 (KxN) tile laid out lane-major: lane l holds 16
// contiguous K elements of column (l&15), K-half selected by l>>4.
// kmajor=0: fragment order [n_tile][k_tile]   (W1: wave walks K for fixed H-tile)
// kmajor=1: fragment order [k_tile][n_tile]   (W2: chunk walks N for fixed K-tile)
// ---------------------------------------------------------------------------
__global__ void pack_weights_bf16(const float* __restrict__ src,
                                  bf16_t* __restrict__ dst,
                                  int K, int Ncols, int kmajor) {
  int gid = blockIdx.x * blockDim.x + threadIdx.x;
  if (gid >= K * Ncols) return;
  int frag   = gid >> 9;            // 512 bf16 per fragment (32 lanes x 16)
  int within = gid & 511;
  int lane   = within >> 4;
  int j      = within & 15;
  int KT = K >> 5, NT = Ncols >> 4;
  int kt, nt;
  if (kmajor) { kt = frag / NT; nt = frag - kt * NT; }
  else        { nt = frag / KT; kt = frag - nt * KT; }
  int k = kt * 32 + ((lane >> 4) << 4) + j;
  int n = nt * 16 + (lane & 15);
  dst[gid] = (bf16_t)src[(size_t)k * Ncols + n];
}

// ---------------------------------------------------------------------------
// Fused main kernel: one workgroup (8 waves / 256 threads) per 32-row slab.
// ---------------------------------------------------------------------------
__global__ __launch_bounds__(256, 1)
void fused_rff_gate_kernel(const float* __restrict__ x,
                           const float* __restrict__ bias,
                           const bf16_t* __restrict__ w1r,
                           const bf16_t* __restrict__ w1s,
                           const bf16_t* __restrict__ w2r,
                           const bf16_t* __restrict__ w2s,
                           float* __restrict__ out) {
  const int tid    = threadIdx.x;
  const int wave   = tid >> 5;
  const int lane   = tid & 31;
  const int l16    = lane & 15;
  const int lhi    = lane >> 4;      // 0/1: K-half and M-half selector
  const int branch = wave >> 2;      // 0 = cos/rff branch, 1 = sigmoid branch
  const int htj    = wave & 3;       // which 16-wide H subtile of the chunk

  __shared__ alignas(64) bf16_t xlds[MTILE * DIN];       // 32 KB, row-major bf16 x tile
  __shared__ alignas(64) bf16_t hbuf[2][MTILE * CHUNK];  // 2 x 4 KB hidden chunk

  const int row0 = blockIdx.x * MTILE;

  // --- load + convert x tile (coalesced float4, once per workgroup) ---
  {
    const float* xg = x + (size_t)row0 * DIN;
#pragma unroll
    for (int i = 0; i < (MTILE * DIN) / 1024; ++i) {     // 16 iters, 1024 floats each
      int idx = (i * 256 + tid) * 4;
      const float4 v = *(const float4*)(xg + idx);
      xlds[idx + 0] = (bf16_t)v.x;
      xlds[idx + 1] = (bf16_t)v.y;
      xlds[idx + 2] = (bf16_t)v.z;
      xlds[idx + 3] = (bf16_t)v.w;
    }
  }
  __syncthreads();

  const v8f vzero = {};
  v8f accR[2][4], accS[2][4];        // 32x512 f32 output tile, both branches
#pragma unroll
  for (int m = 0; m < 2; ++m)
#pragma unroll
    for (int j = 0; j < 4; ++j) { accR[m][j] = vzero; accS[m][j] = vzero; }

  const bf16_t* w1 = branch ? w1s : w1r;

  for (int c = 0; c < NCHUNK; ++c) {
    // ================= stage 1: H_chunk = act(x @ W1[:, chunk]) =================
    const int ht = c * 4 + htj;                       // global 16-wide H tile index
    const bf16_t* wp  = w1 + (size_t)(ht * 16) * 512 + lane * 16;
    const bf16_t* a0p = &xlds[(size_t)l16        * DIN + lhi * 16];
    const bf16_t* a1p = &xlds[(size_t)(l16 + 16) * DIN + lhi * 16];
    v8f h0 = vzero, h1 = vzero;
#pragma unroll
    for (int kt = 0; kt < 16; ++kt) {                 // K = 512 = 16 x 32
      v16bf b  = *(const v16bf*)(wp + (size_t)kt * 512);
      v16bf a0 = *(const v16bf*)(a0p + kt * 32);
      v16bf a1 = *(const v16bf*)(a1p + kt * 32);
      h0 = __builtin_amdgcn_wmma_f32_16x16x32_bf16(false, a0, false, b, (short)0, h0, false, false);
      h1 = __builtin_amdgcn_wmma_f32_16x16x32_bf16(false, a1, false, b, (short)0, h1, false, false);
    }

    // --- activation + bf16 pack into LDS (row-major == A-fragment layout) ---
    const int   ncol = htj * 16 + l16;                // column within chunk [0,64)
    const float bval = bias[c * CHUNK + ncol];
    bf16_t* hdst = &hbuf[branch][0];
#pragma unroll
    for (int r = 0; r < 8; ++r) {
      const int rowA = r + 8 * lhi;                   // C-tile row mapping
      float v0 = h0[r] * S_IN;
      float v1 = h1[r] * S_IN;
      float o0, o1;
      if (branch == 0) {                              // wave-uniform branch
        o0 = __cosf(v0 + bval);
        o1 = __cosf(v1 + bval);
      } else {
        o0 = 1.0f / (1.0f + __expf(-v0));
        o1 = 1.0f / (1.0f + __expf(-v1));
      }
      hdst[(size_t)rowA        * CHUNK + ncol] = (bf16_t)o0;
      hdst[(size_t)(rowA + 16) * CHUNK + ncol] = (bf16_t)o1;
    }
    __syncthreads();

    // prefetch next chunk's W2 fragments into cache (global_prefetch_b8)
    if (c + 1 < NCHUNK) {
      const size_t pf = (size_t)((c + 1) * 64 + wave * 4) * 512;
      __builtin_prefetch(w2r + pf, 0, 1);
      __builtin_prefetch(w2s + pf, 0, 1);
    }

    // ============ stage 2: acc += H_chunk @ W2[chunk, :] (both branches) ============
#pragma unroll
    for (int s = 0; s < 2; ++s) {                     // chunk K = 64 = 2 x 32
      const int kt2 = c * 2 + s;
      v16bf aR[2], aS[2];
#pragma unroll
      for (int m = 0; m < 2; ++m) {
        const int arow = l16 + 16 * m;
        const int kofs = s * 32 + lhi * 16;
        aR[m] = *(const v16bf*)(&hbuf[0][(size_t)arow * CHUNK + kofs]);
        aS[m] = *(const v16bf*)(&hbuf[1][(size_t)arow * CHUNK + kofs]);
      }
#pragma unroll
      for (int j = 0; j < 4; ++j) {
        const int nt = wave * 4 + j;                  // wave owns n-tiles [4w, 4w+4)
        const size_t foff = (size_t)(kt2 * 32 + nt) * 512 + lane * 16;
        v16bf bR = *(const v16bf*)(w2r + foff);
        v16bf bS = *(const v16bf*)(w2s + foff);
#pragma unroll
        for (int m = 0; m < 2; ++m) {
          accR[m][j] = __builtin_amdgcn_wmma_f32_16x16x32_bf16(false, aR[m], false, bR, (short)0, accR[m][j], false, false);
          accS[m][j] = __builtin_amdgcn_wmma_f32_16x16x32_bf16(false, aS[m], false, bS, (short)0, accS[m][j], false, false);
        }
      }
    }
    __syncthreads();   // hbuf is rewritten next iteration
  }

  // ================= epilogue: out = rff * sig * (1/H) =================
#pragma unroll
  for (int m = 0; m < 2; ++m)
#pragma unroll
    for (int j = 0; j < 4; ++j) {
      const int n = (wave * 4 + j) * 16 + l16;
#pragma unroll
      for (int r = 0; r < 8; ++r) {
        const int row = row0 + m * 16 + r + 8 * lhi;
        out[(size_t)row * DOUT + n] = accR[m][j][r] * accS[m][j][r] * S_H2;
      }
    }
}

// ---------------------------------------------------------------------------
extern "C" void kernel_launch(void* const* d_in, const int* in_sizes, int n_in,
                              void* d_out, int out_size, void* d_ws, size_t ws_size,
                              hipStream_t stream) {
  (void)in_sizes; (void)n_in; (void)out_size; (void)ws_size;
  const float* x      = (const float*)d_in[0];
  const float* W_rff1 = (const float*)d_in[1];
  const float* bias   = (const float*)d_in[2];
  const float* W_rff2 = (const float*)d_in[3];
  const float* W_sig1 = (const float*)d_in[4];
  const float* W_sig2 = (const float*)d_in[5];
  float* out = (float*)d_out;

  // workspace: 4 x (512*2048) bf16 packed weight images = 8 MB
  bf16_t* w1r = (bf16_t*)d_ws;
  bf16_t* w1s = w1r + (size_t)DIN * HDIM;
  bf16_t* w2r = w1s + (size_t)DIN * HDIM;
  bf16_t* w2s = w2r + (size_t)HDIM * DOUT;

  const int packBlocks = (DIN * HDIM + 255) / 256;   // 4096
  pack_weights_bf16<<<packBlocks, 256, 0, stream>>>(W_rff1, w1r, DIN,  HDIM, 0);
  pack_weights_bf16<<<packBlocks, 256, 0, stream>>>(W_sig1, w1s, DIN,  HDIM, 0);
  pack_weights_bf16<<<packBlocks, 256, 0, stream>>>(W_rff2, w2r, HDIM, DOUT, 1);
  pack_weights_bf16<<<packBlocks, 256, 0, stream>>>(W_sig2, w2s, HDIM, DOUT, 1);

  const int nBlocks = 65536 / MTILE;                 // 2048 workgroups
  fused_rff_gate_kernel<<<nBlocks, 256, 0, stream>>>(x, bias, w1r, w1s, w2r, w2s, out);
}
